// Attention_17257178595788
// MI455X (gfx1250) — matrix operand order repaired and checked by reading, not compiled, not run on hardware
//
#include <hip/hip_runtime.h>
#include <hip/hip_bf16.h>

// ---------------------------------------------------------------------------
// Types for CDNA5 WMMA (wave32): v_wmma_f32_16x16x32_bf16
// ---------------------------------------------------------------------------
typedef __bf16 bf16_t;
typedef bf16_t v16bf __attribute__((ext_vector_type(16)));
typedef float  v8f   __attribute__((ext_vector_type(8)));

union Frag16 {              // 16 bf16 elements == 8 VGPRs
    v16bf bf;
    uint4 q[2];
};

__device__ __forceinline__ unsigned short f2bf16(float f) {
    unsigned int u = __float_as_uint(f);
    return (unsigned short)((u + 0x7FFFu + ((u >> 16) & 1u)) >> 16);
}
__device__ __forceinline__ unsigned int pack_bf16x2(float lo, float hi) {
    unsigned int ul = __float_as_uint(lo);
    unsigned int uh = __float_as_uint(hi);
    ul = (ul + 0x7FFFu + ((ul >> 16) & 1u)) >> 16;
    uh = (uh + 0x7FFFu + ((uh >> 16) & 1u)) >> 16;
    return (uh << 16) | (ul & 0xFFFFu);
}

__device__ __forceinline__ v8f wmma_bf16(const Frag16& a, const Frag16& b, v8f c) {
    return __builtin_amdgcn_wmma_f32_16x16x32_bf16(
        /*neg_a=*/false, a.bf, /*neg_b=*/false, b.bf,
        /*c_mod=*/(short)0, c, /*reuse_a=*/false, /*reuse_b=*/false);
}

// CDNA5 async copy: 16 bytes global -> LDS, tracked on ASYNCcnt.
__device__ __forceinline__ void async_copy16(const void* gptr, void* lptr) {
    unsigned long long ga = (unsigned long long)gptr;
    unsigned int       la = (unsigned int)(unsigned long long)lptr; // addr[31:0] = LDS offset
    asm volatile("global_load_async_to_lds_b128 %0, %1, off"
                 :: "v"(la), "v"(ga) : "memory");
}
__device__ __forceinline__ void wait_async0() {
    asm volatile("s_wait_asynccnt 0" ::: "memory");
}

// ---------------------------------------------------------------------------
// GEMM, block tile 128x128x32, 256 threads (8 waves), wave tile 32x64.
// LDS: A as [m][k] bf16; B as [n][k] bf16 (K-transposed) -> WMMA fragments are
// contiguous ds_load_b128 per the CDNA5 lane layouts.
// MODE 0: A = f32 (x), outputs bf16, split:  cols [0,1536) -> qk[m][1536]
//         (Q columns [0,768) pre-scaled by 1/sqrt(Hd) so the flash kernel's
//          S needs no post-WMMA multiply), cols [1536,2304) ->
//         vT[(b*H+h)*64+d][token]  (V pre-transposed).
// MODE 1: A = bf16 (async-copied to LDS), output f32 + bias (projection).
// ---------------------------------------------------------------------------
#define GBM 128
#define GBN 128
#define GBK 32

template<int MODE>
__global__ __launch_bounds__(256)
void gemm_kernel(const void* __restrict__ Av, const float* __restrict__ B,
                 const float* __restrict__ bias,
                 void* __restrict__ out0, unsigned short* __restrict__ vT,
                 int M, int N, int K, int Hn, int Nn, float qscale) {
    __shared__ __align__(16) unsigned short sA[GBM * GBK]; // [m][k]
    __shared__ __align__(16) unsigned short sB[GBN * GBK]; // [n][k]

    const int tid  = threadIdx.x;
    const int lane = tid & 31;
    const int wave = tid >> 5;      // 0..7
    const int wm   = wave >> 1;     // 0..3  (row group of 32)
    const int wn   = wave & 1;      // 0..1  (col group of 64)
    const int hl   = lane >> 4;     // half-wave id
    const int l15  = lane & 15;

    const int m0 = blockIdx.y * GBM;
    const int n0 = blockIdx.x * GBN;

    v8f acc[2][4];
    #pragma unroll
    for (int i = 0; i < 2; ++i)
        #pragma unroll
        for (int j = 0; j < 4; ++j)
            acc[i][j] = (v8f){0.f,0.f,0.f,0.f,0.f,0.f,0.f,0.f};

    for (int k0 = 0; k0 < K; k0 += GBK) {
        if (MODE == 0) {
            // A tile f32 -> bf16: 128 rows x 32 k, 1024 float4, 4 per thread
            const float* A = (const float*)Av;
            #pragma unroll
            for (int i = 0; i < 4; ++i) {
                int j   = tid + 256 * i;
                int row = j >> 3;
                int c4  = j & 7;
                const float4 f = *reinterpret_cast<const float4*>(
                    A + (size_t)(m0 + row) * K + k0 + c4 * 4);
                uint2 u;
                u.x = pack_bf16x2(f.x, f.y);
                u.y = pack_bf16x2(f.z, f.w);
                *reinterpret_cast<uint2*>(&sA[row * GBK + c4 * 4]) = u;
            }
        } else {
            // A tile already bf16: raw async copy, 512 x 16B chunks, 2/thread
            const unsigned short* A = (const unsigned short*)Av;
            #pragma unroll
            for (int i = 0; i < 2; ++i) {
                int j   = tid + 256 * i;   // 0..511
                int row = j >> 2;          // 0..127 (4 chunks per 32-half row)
                int c8  = j & 3;
                async_copy16(A + (size_t)(m0 + row) * K + k0 + c8 * 8,
                             &sA[row * GBK + c8 * 8]);
            }
        }
        // B tile f32 -> bf16, transposed to sB[n][k]: 8 k-pair x 128 n / thread
        #pragma unroll
        for (int i = 0; i < 8; ++i) {
            int j  = tid + 256 * i;    // 0..2047
            int k2 = j >> 7;           // 0..15 : pair of k rows
            int n  = j & 127;
            const float b0 = B[(size_t)(k0 + 2 * k2 + 0) * N + n0 + n];
            const float b1 = B[(size_t)(k0 + 2 * k2 + 1) * N + n0 + n];
            *reinterpret_cast<unsigned int*>(&sB[n * GBK + 2 * k2]) =
                pack_bf16x2(b0, b1);
        }
        if (MODE == 1) wait_async0();
        __syncthreads();

        Frag16 af[2];
        #pragma unroll
        for (int mt = 0; mt < 2; ++mt) {
            int row = wm * 32 + mt * 16 + l15;
            af[mt].q[0] = *reinterpret_cast<const uint4*>(&sA[row * GBK + hl * 8]);
            af[mt].q[1] = *reinterpret_cast<const uint4*>(&sA[row * GBK + 16 + hl * 8]);
        }
        #pragma unroll
        for (int nt = 0; nt < 4; ++nt) {
            Frag16 bf;
            int n = wn * 64 + nt * 16 + l15;
            const uint4* p = reinterpret_cast<const uint4*>(&sB[n * GBK + hl * 8]);
            bf.q[0] = p[0];
            bf.q[1] = p[2];
            #pragma unroll
            for (int mt = 0; mt < 2; ++mt)
                acc[mt][nt] = wmma_bf16(af[mt], bf, acc[mt][nt]);
        }
        __syncthreads();
    }

    // Epilogue. Lane holds column l15; VGPR v holds row (v + 8*hl).
    #pragma unroll
    for (int mt = 0; mt < 2; ++mt) {
        #pragma unroll
        for (int nt = 0; nt < 4; ++nt) {
            int gm = m0 + wm * 32 + mt * 16 + hl * 8;
            int gn = n0 + wn * 64 + nt * 16 + l15;
            if (MODE == 1) {
                float* C = (float*)out0;
                float bv = bias[gn];
                #pragma unroll
                for (int v = 0; v < 8; ++v)
                    C[(size_t)(gm + v) * N + gn] = acc[mt][nt][v] + bv;
            } else {
                unsigned short* qk = (unsigned short*)out0;
                const int QW  = Hn * 64;                // 768
                const int QKW = 2 * QW;                 // 1536
                if (gn < QKW) {
                    float sc = (gn < QW) ? qscale : 1.0f;   // fold 1/sqrt(Hd) into Q
                    #pragma unroll
                    for (int v = 0; v < 8; ++v)
                        qk[(size_t)(gm + v) * QKW + gn] = f2bf16(acc[mt][nt][v] * sc);
                } else {
                    int c  = gn - QKW;
                    int hh = c >> 6;
                    int dd = c & 63;
                    #pragma unroll
                    for (int v = 0; v < 8; ++v) {
                        int gr  = gm + v;
                        int bb  = gr >> 12;             // token row / 4096
                        int tok = gr & 4095;
                        vT[(size_t)((bb * Hn + hh) * 64 + dd) * Nn + tok] =
                            f2bf16(acc[mt][nt][v]);
                    }
                }
            }
        }
    }
}

// ---------------------------------------------------------------------------
// Flash attention, bf16 in / bf16 out. 4 waves per (b, h, 64-query block);
// each wave owns 16 query rows. Key block Bc=128 (32 WMMAs between barriers).
// K tile async-copied to sK[n][d]; V tile (pre-transposed) to sV[d][n].
// Q is pre-scaled, so S = WMMA output directly feeds softmax.
// ---------------------------------------------------------------------------
#define ABR 64
#define ABC 128
#define AHD 64

__global__ __launch_bounds__(128)
void flash_attn_kernel(const unsigned short* __restrict__ qk,
                       const unsigned short* __restrict__ vT,
                       unsigned short* __restrict__ out,
                       int Bn, int Nn, int Hn) {
    __shared__ __align__(16) unsigned short sK[ABC * AHD];      // [n][d] 16 KB
    __shared__ __align__(16) unsigned short sV[AHD * ABC];      // [d][n] 16 KB
    __shared__ __align__(16) unsigned short sP[4][16 * ABC];    // 16 KB

    const int tid  = threadIdx.x;
    const int lane = tid & 31;
    const int wave = tid >> 5;
    const int hl   = lane >> 4;
    const int l15  = lane & 15;

    const int q0 = blockIdx.x * ABR;
    const int h  = blockIdx.y;
    const int b  = blockIdx.z;
    const int QKW  = 2 * Hn * AHD;          // 1536 halves per token row
    const int kCol = Hn * AHD + h * AHD;    // K column offset in qk row

    // --- Q fragments (bf16, direct 16B loads in the A-fragment layout) ---
    Frag16 qf[2];
    {
        int m = wave * 16 + l15;
        const unsigned short* qrow = qk + (size_t)(b * Nn + q0 + m) * QKW + h * AHD;
        #pragma unroll
        for (int c = 0; c < 2; ++c) {
            qf[c].q[0] = *reinterpret_cast<const uint4*>(qrow + c * 32 + hl * 8);
            qf[c].q[1] = *reinterpret_cast<const uint4*>(qrow + c * 32 + 16 + hl * 8);
        }
    }

    float mrun[8], lrun[8];
    v8f o[4];
    #pragma unroll
    for (int v = 0; v < 8; ++v) { mrun[v] = -1e30f; lrun[v] = 0.f; }
    #pragma unroll
    for (int t = 0; t < 4; ++t)
        o[t] = (v8f){0.f,0.f,0.f,0.f,0.f,0.f,0.f,0.f};

    const unsigned short* vBase = vT + (size_t)((b * Hn + h) * AHD) * Nn;

    for (int kv0 = 0; kv0 < Nn; kv0 += ABC) {
        // --- async copy K tile: 128 rows x 128B -> sK[n][d], 8 chunks/thread
        #pragma unroll
        for (int i = 0; i < 8; ++i) {
            int j  = tid + 128 * i;    // 0..1023 16B chunks
            int n  = j >> 3;
            int c8 = j & 7;
            async_copy16(qk + (size_t)(b * Nn + kv0 + n) * QKW + kCol + c8 * 8,
                         &sK[n * AHD + c8 * 8]);
        }
        // --- async copy V tile (pre-transposed): 64 d-rows x 256B -> sV[d][n]
        #pragma unroll
        for (int i = 0; i < 8; ++i) {
            int j  = tid + 128 * i;    // 0..1023
            int d  = j >> 4;
            int c8 = j & 15;
            async_copy16(vBase + (size_t)d * Nn + kv0 + c8 * 8,
                         &sV[d * ABC + c8 * 8]);
        }
        // --- prefetch next K tile rows while copies are in flight ---
        if (kv0 + ABC < Nn) {
            __builtin_prefetch(qk + (size_t)(b * Nn + kv0 + ABC + tid) * QKW + kCol,
                               0, 1);
        }
        wait_async0();
        __syncthreads();

        // --- S = Q @ K^T : 8 tiles of 16 key columns (Q already scaled) ---
        v8f s[8];
        #pragma unroll
        for (int nt = 0; nt < 8; ++nt) {
            s[nt] = (v8f){0.f,0.f,0.f,0.f,0.f,0.f,0.f,0.f};
            #pragma unroll
            for (int c = 0; c < 2; ++c) {
                Frag16 kf;
                int n = nt * 16 + l15;
                const uint4* p = reinterpret_cast<const uint4*>(
                    &sK[n * AHD + c * 32 + hl * 8]);
                kf.q[0] = p[0];
                kf.q[1] = p[2];
                s[nt] = wmma_bf16(qf[c], kf, s[nt]);
            }
        }

        // --- online softmax; write P (bf16) into this wave's sP tile ---
        #pragma unroll
        for (int v = 0; v < 8; ++v) {
            float mx = s[0][v];
            #pragma unroll
            for (int nt = 1; nt < 8; ++nt)
                mx = fmaxf(mx, s[nt][v]);
            #pragma unroll
            for (int off = 8; off >= 1; off >>= 1)
                mx = fmaxf(mx, __shfl_xor(mx, off, 32));
            float newm = fmaxf(mrun[v], mx);
            float corr = __expf(mrun[v] - newm);
            mrun[v] = newm;
            float p[8];
            float rs = 0.f;
            #pragma unroll
            for (int nt = 0; nt < 8; ++nt) {
                p[nt] = __expf(s[nt][v] - newm);
                rs += p[nt];
            }
            #pragma unroll
            for (int off = 8; off >= 1; off >>= 1)
                rs += __shfl_xor(rs, off, 32);
            lrun[v] = lrun[v] * corr + rs;
            #pragma unroll
            for (int ot = 0; ot < 4; ++ot)
                o[ot][v] *= corr;
            int m = v + 8 * hl;
            #pragma unroll
            for (int nt = 0; nt < 8; ++nt)
                sP[wave][m * ABC + nt * 16 + l15] = f2bf16(p[nt]);
        }

        // --- O += P @ V : 4 K-chunks of 32 keys x 4 d-tiles ---
        #pragma unroll
        for (int c = 0; c < 4; ++c) {
            Frag16 pf;
            pf.q[0] = *reinterpret_cast<const uint4*>(
                &sP[wave][l15 * ABC + c * 32 + hl * 8]);
            pf.q[1] = *reinterpret_cast<const uint4*>(
                &sP[wave][l15 * ABC + c * 32 + 16 + hl * 8]);
            #pragma unroll
            for (int ot = 0; ot < 4; ++ot) {
                Frag16 vf;
                int d = ot * 16 + l15;
                const uint4* vp = reinterpret_cast<const uint4*>(
                    &sV[d * ABC + c * 32 + hl * 8]);
                vf.q[0] = vp[0];
                vf.q[1] = vp[2];
                o[ot] = wmma_bf16(pf, vf, o[ot]);
            }
        }
        __syncthreads();
    }

    // --- finalize: divide by row sums, store bf16 [b, n, h*64+d] ---
    const int Dout = Hn * AHD;
    #pragma unroll
    for (int v = 0; v < 8; ++v) {
        float inv = 1.0f / lrun[v];
        int m = q0 + wave * 16 + v + 8 * hl;
        #pragma unroll
        for (int ot = 0; ot < 4; ++ot) {
            int d = ot * 16 + l15;
            out[(size_t)(b * Nn + m) * Dout + h * AHD + d] = f2bf16(o[ot][v] * inv);
        }
    }
}

// ---------------------------------------------------------------------------
// Launch: qkv GEMM (bf16 out, Q pre-scaled, V pre-transposed) ->
// flash attention (bf16) -> projection GEMM (bf16 A via async copy, f32+bias)
// ---------------------------------------------------------------------------
extern "C" void kernel_launch(void* const* d_in, const int* in_sizes, int n_in,
                              void* d_out, int out_size, void* d_ws, size_t ws_size,
                              hipStream_t stream) {
    const float* x      = (const float*)d_in[0];
    const float* w_qkv  = (const float*)d_in[1];
    const float* w_proj = (const float*)d_in[2];
    const float* b_proj = (const float*)d_in[3];
    float*       out    = (float*)d_out;

    const int Bsz = 2, N = 4096, D = 768, H = 12;
    const int M = Bsz * N;                          // 8192 token rows

    unsigned short* qk    = (unsigned short*)d_ws;          // [M, 1536] bf16
    unsigned short* vT    = qk + (size_t)M * 2 * D;         // [B*H*64, N] bf16
    unsigned short* attnB = vT + (size_t)Bsz * H * 64 * N;  // [M, 768]  bf16

    // 1) qkv = x @ w_qkv  (bf16 out; Q scaled by 1/8, Q/K compact, V transposed)
    {
        dim3 block(256);
        dim3 grid((3 * D) / GBN, M / GBM);          // 18 x 64
        gemm_kernel<0><<<grid, block, 0, stream>>>(
            x, w_qkv, nullptr, qk, vT, M, 3 * D, D, H, N, 0.125f);
    }
    // 2) flash attention per (b, h, 64-query block)
    {
        dim3 block(128);
        dim3 grid(N / ABR, H, Bsz);                 // 64 x 12 x 2
        flash_attn_kernel<<<grid, block, 0, stream>>>(
            qk, vT, attnB, Bsz, N, H);
    }
    // 3) out = attn @ w_proj + b_proj  (A bf16 via async copy)
    {
        dim3 block(256);
        dim3 grid(D / GBN, M / GBM);                // 6 x 64
        gemm_kernel<1><<<grid, block, 0, stream>>>(
            attnB, w_proj, b_proj, out, nullptr, M, D, D, H, N, 1.0f);
    }
}
